// MultiHeadAttention_89687507076070
// MI455X (gfx1250) — compile-verified, matrix operand order checked by reading, not run on hardware
//
#include <hip/hip_runtime.h>
#include <hip/hip_bf16.h>
#include <cstdint>

// ---------------------------------------------------------------------------
// Linear attention (silu+1 feature map), MI455X / gfx1250, wave32 + WMMA.
// Pipeline: bf16 convert -> 3 projection GEMMs (phi/scale/bias epilogues)
//           -> batched kv GEMM -> batched out GEMM -> final projection (f32).
// GEMM core: D = A * B^T, K-contiguous operands, double-buffered LDS fed by
// CDNA5 async global->LDS copies (ASYNCcnt), math on v_wmma_f32_16x16x32_bf16.
// ---------------------------------------------------------------------------

typedef __attribute__((ext_vector_type(8)))  __bf16 v8bf;
typedef __attribute__((ext_vector_type(16))) __bf16 v16bf;
typedef __attribute__((ext_vector_type(4)))  __bf16 v4bf;
typedef __attribute__((ext_vector_type(8)))  float  v8f;
typedef __attribute__((__vector_size__(4 * sizeof(int)))) int v4i;

#define B_BATCH 8
#define T_SEQ   4096
#define D_MODEL 1024
#define N_HEAD  16
#define DH      64

static constexpr float QK_SCALE = 0.35355339059327373f; // 64^-0.25

enum { L_HEAD_TDh = 0, L_HEAD_DhT = 1, L_BATCH_TRANS = 2, L_BATCH_MERGE = 3, L_F32_BIAS = 4 };

// ---------------- CDNA5 async global->LDS copy (16B per lane) --------------
__device__ __forceinline__ void async_cp16(const void* g, void* l) {
#if __has_builtin(__builtin_amdgcn_global_load_async_to_lds_b128)
  __builtin_amdgcn_global_load_async_to_lds_b128(
      (__attribute__((address_space(1))) v4i*)g,
      (__attribute__((address_space(3))) v4i*)l, 0, 0);
#else
  unsigned int laddr = (unsigned int)(uintptr_t)(__attribute__((address_space(3))) void*)l;
  unsigned long long gaddr = (unsigned long long)(uintptr_t)g;
  asm volatile("global_load_async_to_lds_b128 %0, %1, off"
               :: "v"(laddr), "v"(gaddr) : "memory");
#endif
}

template <int N>
__device__ __forceinline__ void wait_async() {
#if __has_builtin(__builtin_amdgcn_s_wait_asynccnt)
  __builtin_amdgcn_s_wait_asynccnt((unsigned short)N);
#else
  asm volatile("s_wait_asynccnt %0" :: "n"(N) : "memory");
#endif
}

// -------------------------- f32 -> bf16 convert ----------------------------
__global__ __launch_bounds__(256) void cvt_f32_bf16(const float* __restrict__ in,
                                                    unsigned short* __restrict__ out,
                                                    int n) {
  int i = (blockIdx.x * 256 + threadIdx.x) * 4;
  if (i + 3 < n) {
    float4 f = *(const float4*)(in + i);
    v4bf o;
    o.x = (__bf16)f.x; o.y = (__bf16)f.y; o.z = (__bf16)f.z; o.w = (__bf16)f.w;
    *(v4bf*)((__bf16*)out + i) = o;
  } else {
    for (; i < n; ++i) ((__bf16*)out)[i] = (__bf16)in[i];
  }
}

// ------------------------------ WMMA GEMM ----------------------------------
// Block: 256 threads (8 wave32). Tile: BM x 64 output, K stepped by 32.
// Wave grid: 4 waves along M x 2 along N; each wave MF x 2 WMMA tiles.
// LDS: double-buffered [A tile BM x 32 | B tile 64 x 32], pitch 40 halves.
template <int BM, int LAYOUT, bool PHI, bool HASBIAS>
__global__ __launch_bounds__(256) void gemm_bf16_wmma(
    const unsigned short* __restrict__ Au, const unsigned short* __restrict__ Bu,
    const float* __restrict__ bias, void* __restrict__ outp,
    int M, int N, int K, int lda, int ldb,
    unsigned long long strideA, unsigned long long strideB) {
  constexpr int APITCH = 40;                 // 32 bf16 + 8 pad -> conflict-free b128
  constexpr int MF     = (BM == 128) ? 2 : 1;
  constexpr int ACH    = BM * 4;             // # of 16B chunks in the A tile
  constexpr int CH     = (ACH + 256) / 256;  // async copies per thread per tile
  constexpr int ASIZE  = BM * APITCH;        // A region (elements)
  constexpr int TSIZE  = ASIZE + 64 * APITCH;

  __shared__ __align__(16) __bf16 lds[2][TSIZE];

  const __bf16* A  = (const __bf16*)Au + (size_t)blockIdx.z * strideA;
  const __bf16* Bm = (const __bf16*)Bu + (size_t)blockIdx.z * strideB;

  const int tid  = threadIdx.x;
  const int lane = tid & 31;
  const int wid  = tid >> 5;
  const int half = lane >> 4;
  const int l16  = lane & 15;

  const int blockM = blockIdx.y * BM;
  const int blockN = blockIdx.x * 64;
  const int waveM  = (wid & 3) * (MF * 16);
  const int waveN  = (wid >> 2) * 32;

  // Per-thread staging chunks: q*256 < ACH -> A region, else B region.
  const __bf16* gsrc[CH];
  unsigned int  loff[CH];
#pragma unroll
  for (int q = 0; q < CH; ++q) {
    if (q * 256 < ACH) {
      int c = tid + q * 256, row = c >> 2, col = (c & 3) << 3;
      loff[q] = row * APITCH + col;
      gsrc[q] = A + (size_t)(blockM + row) * lda + col;
    } else {
      int c = tid + q * 256 - ACH, row = c >> 2, col = (c & 3) << 3;
      loff[q] = ASIZE + row * APITCH + col;
      gsrc[q] = Bm + (size_t)(blockN + row) * ldb + col;
    }
  }

  v8f acc[MF][2];
#pragma unroll
  for (int i = 0; i < MF; ++i)
#pragma unroll
    for (int j = 0; j < 2; ++j) acc[i][j] = {};

  const int nIter = K >> 5;
  // Prologue: stage tile 0 into buffer 0.
#pragma unroll
  for (int q = 0; q < CH; ++q) async_cp16(gsrc[q], &lds[0][loff[q]]);

  for (int it = 0; it < nIter; ++it) {
    const int cur = it & 1;
    if (it + 1 < nIter) {  // prefetch next tile into the other buffer
      const int k0 = (it + 1) << 5;
#pragma unroll
      for (int q = 0; q < CH; ++q) async_cp16(gsrc[q] + k0, &lds[cur ^ 1][loff[q]]);
      wait_async<CH>();    // loads retire in order: tile `it` is now resident
    } else {
      wait_async<0>();
    }
    __syncthreads();       // tile `it` visible to all waves

    const __bf16* bufA = &lds[cur][0];
    const __bf16* bufB = &lds[cur][ASIZE];

    // Fragment loads per documented 16-bit A/B layouts: lane l16 = row,
    // half selects K ranges {8h..8h+7} and {16+8h..23+8h} -> two 16B reads.
    v16bf afrag[MF], bfrag[2];
#pragma unroll
    for (int i = 0; i < MF; ++i) {
      const __bf16* rp = bufA + (waveM + i * 16 + l16) * APITCH;
      v8bf lo = *(const v8bf*)(rp + half * 8);
      v8bf hi = *(const v8bf*)(rp + 16 + half * 8);
      afrag[i] = __builtin_shufflevector(lo, hi, 0, 1, 2, 3, 4, 5, 6, 7, 8, 9, 10, 11, 12, 13, 14, 15);
    }
#pragma unroll
    for (int j = 0; j < 2; ++j) {
      const __bf16* rp = bufB + (waveN + j * 16 + l16) * APITCH;
      v8bf lo = *(const v8bf*)(rp + half * 8);
      v8bf hi = *(const v8bf*)(rp + 16 + half * 8);
      bfrag[j] = __builtin_shufflevector(lo, hi, 0, 1, 2, 3, 4, 5, 6, 7, 8, 9, 10, 11, 12, 13, 14, 15);
    }

#pragma unroll
    for (int i = 0; i < MF; ++i)
#pragma unroll
      for (int j = 0; j < 2; ++j)
        acc[i][j] = __builtin_amdgcn_wmma_f32_16x16x32_bf16(
            false, afrag[i], false, bfrag[j], (short)0, acc[i][j], false, false);

    __syncthreads();       // all waves done reading buf `cur`; safe to refill
  }

  // Epilogue. C/D layout: VGPR r <-> m = r + 8*half, n = l16.
#pragma unroll
  for (int j = 0; j < 2; ++j) {
    const int gn = blockN + waveN + j * 16 + l16;
    float bval = 0.f;
    if (HASBIAS) bval = bias[gn];
#pragma unroll
    for (int i = 0; i < MF; ++i) {
#pragma unroll
      for (int r = 0; r < 8; ++r) {
        const int gm = blockM + waveM + i * 16 + (half << 3) + r;
        float v = acc[i][j][r];
        if (HASBIAS) v += bval;
        if (PHI) { v *= QK_SCALE; v = v / (1.f + __expf(-v)) + 1.f; }  // silu(v)+1
        if (LAYOUT == L_HEAD_TDh) {
          int b = gm >> 12, t = gm & (T_SEQ - 1), h = gn >> 6, e = gn & (DH - 1);
          ((__bf16*)outp)[((size_t)((b << 4) + h) * T_SEQ + t) * DH + e] = (__bf16)v;
        } else if (LAYOUT == L_HEAD_DhT) {
          int b = gm >> 12, t = gm & (T_SEQ - 1), h = gn >> 6, e = gn & (DH - 1);
          ((__bf16*)outp)[((size_t)((b << 4) + h) * DH + e) * T_SEQ + t] = (__bf16)v;
        } else if (LAYOUT == L_BATCH_TRANS) {     // write kv^T : [e][d]
          ((__bf16*)outp)[(size_t)blockIdx.z * (DH * DH) + (size_t)gn * DH + gm] = (__bf16)v;
        } else if (LAYOUT == L_BATCH_MERGE) {     // heads back to [B,T,D]
          int b = blockIdx.z >> 4, h = blockIdx.z & 15;
          ((__bf16*)outp)[(size_t)b * T_SEQ * D_MODEL + (size_t)gm * D_MODEL + h * DH + gn] = (__bf16)v;
        } else {                                  // L_F32_BIAS : final output
          ((float*)outp)[(size_t)gm * D_MODEL + gn] = v;
        }
      }
    }
  }
}

// ------------------------------- launcher ----------------------------------
extern "C" void kernel_launch(void* const* d_in, const int* in_sizes, int n_in,
                              void* d_out, int out_size, void* d_ws, size_t ws_size,
                              hipStream_t stream) {
  (void)in_sizes; (void)n_in; (void)out_size; (void)ws_size;
  const float* x  = (const float*)d_in[0];
  const float* Wq = (const float*)d_in[1];
  const float* bq = (const float*)d_in[2];
  const float* Wk = (const float*)d_in[3];
  const float* Wv = (const float*)d_in[4];
  const float* bv = (const float*)d_in[5];
  const float* Wo = (const float*)d_in[6];
  const float* bo = (const float*)d_in[7];
  float* out = (float*)d_out;

  const size_t MTOT = (size_t)B_BATCH * T_SEQ;      // 32768
  const size_t XEL  = MTOT * D_MODEL;               // 33.5M
  const size_t WEL  = (size_t)D_MODEL * D_MODEL;    // 1M

  char* ws = (char*)d_ws;
  auto alloc = [&](size_t bytes) {
    void* p = (void*)ws;
    ws += (bytes + 255) & ~(size_t)255;
    return p;
  };
  unsigned short* xbf   = (unsigned short*)alloc(XEL * 2);
  unsigned short* wqb   = (unsigned short*)alloc(WEL * 2);
  unsigned short* wkb   = (unsigned short*)alloc(WEL * 2);
  unsigned short* wvb   = (unsigned short*)alloc(WEL * 2);
  unsigned short* wob   = (unsigned short*)alloc(WEL * 2);
  unsigned short* phiQ  = (unsigned short*)alloc(XEL * 2);  // [B,H,T,Dh]
  unsigned short* phiKT = (unsigned short*)alloc(XEL * 2);  // [B,H,Dh,T]
  unsigned short* VT    = (unsigned short*)alloc(XEL * 2);  // [B,H,Dh,T]
  unsigned short* kvT   = (unsigned short*)alloc((size_t)B_BATCH * N_HEAD * DH * DH * 2);
  unsigned short* wvbuf = (unsigned short*)alloc(XEL * 2);  // [B,T,D]

  dim3 blk(256, 1, 1);
  // converts
  cvt_f32_bf16<<<(int)((XEL / 4 + 255) / 256), blk, 0, stream>>>(x, xbf, (int)XEL);
  cvt_f32_bf16<<<(int)((WEL / 4 + 255) / 256), blk, 0, stream>>>(Wq, wqb, (int)WEL);
  cvt_f32_bf16<<<(int)((WEL / 4 + 255) / 256), blk, 0, stream>>>(Wk, wkb, (int)WEL);
  cvt_f32_bf16<<<(int)((WEL / 4 + 255) / 256), blk, 0, stream>>>(Wv, wvb, (int)WEL);
  cvt_f32_bf16<<<(int)((WEL / 4 + 255) / 256), blk, 0, stream>>>(Wo, wob, (int)WEL);

  // Projections: M=32768, N=1024, K=1024, grid = (N/64, M/128, 1)
  dim3 gProj(D_MODEL / 64, (unsigned)(MTOT / 128), 1);
  gemm_bf16_wmma<128, L_HEAD_TDh, true, true><<<gProj, blk, 0, stream>>>(
      xbf, wqb, bq, phiQ, (int)MTOT, D_MODEL, D_MODEL, D_MODEL, D_MODEL, 0ull, 0ull);
  gemm_bf16_wmma<128, L_HEAD_DhT, true, false><<<gProj, blk, 0, stream>>>(
      xbf, wkb, nullptr, phiKT, (int)MTOT, D_MODEL, D_MODEL, D_MODEL, D_MODEL, 0ull, 0ull);
  gemm_bf16_wmma<128, L_HEAD_DhT, false, true><<<gProj, blk, 0, stream>>>(
      xbf, wvb, bv, VT, (int)MTOT, D_MODEL, D_MODEL, D_MODEL, D_MODEL, 0ull, 0ull);

  // kv = phi(k)^T v : per (b,h) 64x64x4096, grid = (1,1,128), write kv^T bf16
  dim3 gKV(1, 1, B_BATCH * N_HEAD);
  gemm_bf16_wmma<64, L_BATCH_TRANS, false, false><<<gKV, blk, 0, stream>>>(
      phiKT, VT, nullptr, kvT, DH, DH, T_SEQ, T_SEQ, T_SEQ,
      (unsigned long long)DH * T_SEQ, (unsigned long long)DH * T_SEQ);

  // out = phi(q) kv : per (b,h) 4096x64x64, grid = (1, 32, 128), merge heads
  dim3 gOut(1, T_SEQ / 128, B_BATCH * N_HEAD);
  gemm_bf16_wmma<128, L_BATCH_MERGE, false, false><<<gOut, blk, 0, stream>>>(
      phiQ, kvT, nullptr, wvbuf, T_SEQ, DH, DH, DH, DH,
      (unsigned long long)T_SEQ * DH, (unsigned long long)DH * DH);

  // final projection: wv @ Wo^T + bo -> f32 out
  gemm_bf16_wmma<128, L_F32_BIAS, false, true><<<gProj, blk, 0, stream>>>(
      wvbuf, wob, bo, out, (int)MTOT, D_MODEL, D_MODEL, D_MODEL, D_MODEL, 0ull, 0ull);
}
